// NTXent_18708877541981
// MI455X (gfx1250) — matrix operand order repaired and checked by reading, not compiled, not run on hardware
//
#include <hip/hip_runtime.h>
#include <hip/hip_bf16.h>
#include <math.h>

// ---------------------------------------------------------------------------
// NT-Xent loss, tau = 0.5.  B = 4096, D = 512, N = 2B = 8192.
//   z = l2norm(rows of [e_i; e_j])                (f16 in workspace)
//   denom[n] = sum_{m != n} exp(2 * <z_n, z_m>)   (WMMA f16 Gram, symmetric:
//              only upper-triangular 128x128 tiles computed; off-diagonal
//              tiles feed both row sums and column sums)
//   S_pos    = sum_b <z_i_b, z_j_b>               (exact f32 from inputs)
//   loss     = (sum_n log(denom[n]) - 4*S_pos) / 8192
// ---------------------------------------------------------------------------

#define DIM   512
#define BROWS 4096
#define NROWS 8192   // 2B

#define KC    32               // K-halves per LDS stage
#define LROW  40               // padded LDS row stride (halves): 32 data + 8 pad
#define CH    (128 * LROW)     // halves per chunk (A or B) = 5120 (10 KB)

typedef __attribute__((ext_vector_type(16))) _Float16 v16h;
typedef __attribute__((ext_vector_type(8)))  _Float16 v8h;
typedef __attribute__((ext_vector_type(8)))  float    v8f;

// -------------------------- block reduction helper -------------------------
__device__ inline float block_reduce_sum(float v, float* smem) {
    const int lane = threadIdx.x & 31;
    const int wid  = threadIdx.x >> 5;
    #pragma unroll
    for (int o = 16; o > 0; o >>= 1) v += __shfl_down(v, o, 32);
    if (lane == 0) smem[wid] = v;
    __syncthreads();
    v = (threadIdx.x < 8) ? smem[threadIdx.x] : 0.0f;
    if (wid == 0) {
        #pragma unroll
        for (int o = 4; o > 0; o >>= 1) v += __shfl_down(v, o, 32);
    }
    __syncthreads();
    return v;                  // valid in thread 0 only
}

// -------------------------- kernel 0: zero scratch -------------------------
__global__ void ntx_init_kernel(float* __restrict__ denom_and_pos) {
    int i = blockIdx.x * blockDim.x + threadIdx.x;
    if (i < NROWS + 1) denom_and_pos[i] = 0.0f;   // denom[0..8191], S_pos at [8192]
}

// ------------------- kernel 1: l2-normalize -> f16 Z rows -------------------
__global__ __launch_bounds__(256) void ntx_normalize_kernel(
        const float* __restrict__ ei, const float* __restrict__ ej,
        _Float16* __restrict__ z) {
    __shared__ float smem[8];
    __shared__ float s_inv;
    const int row = blockIdx.x;                       // 0..8191
    const float* src = (row < BROWS) ? (ei + (size_t)row * DIM)
                                     : (ej + (size_t)(row - BROWS) * DIM);
    float x0 = src[threadIdx.x];
    float x1 = src[threadIdx.x + 256];
    float ss = block_reduce_sum(x0 * x0 + x1 * x1, smem);
    if (threadIdx.x == 0) s_inv = 1.0f / fmaxf(sqrtf(ss), 1e-12f);
    __syncthreads();
    const float inv = s_inv;
    _Float16* dst = z + (size_t)row * DIM;
    dst[threadIdx.x]       = (_Float16)(x0 * inv);
    dst[threadIdx.x + 256] = (_Float16)(x1 * inv);
}

// ------------- kernel 2: exact positive-pair similarity sum ----------------
__global__ __launch_bounds__(256) void ntx_pos_kernel(
        const float* __restrict__ ei, const float* __restrict__ ej,
        float* __restrict__ spos) {
    __shared__ float smem[8];
    const int b = blockIdx.x;                          // 0..4095
    const float* pi = ei + (size_t)b * DIM;
    const float* pj = ej + (size_t)b * DIM;
    float sii = 0.0f, sjj = 0.0f, sij = 0.0f;
    #pragma unroll
    for (int k = 0; k < 2; ++k) {
        const int i = threadIdx.x + k * 256;
        const float x = pi[i], y = pj[i];
        sii += x * x; sjj += y * y; sij += x * y;
    }
    const float rii = block_reduce_sum(sii, smem);
    const float rjj = block_reduce_sum(sjj, smem);
    const float rij = block_reduce_sum(sij, smem);
    if (threadIdx.x == 0) {
        const float d = fmaxf(sqrtf(rii), 1e-12f) * fmaxf(sqrtf(rjj), 1e-12f);
        atomicAdd(spos, rij / d);
    }
}

// -------------------------- LDS helpers ------------------------------------
__device__ __forceinline__ void ntx_store16(_Float16* p, v16h v) {
    // 16B-aligned pieces (LROW stride gives only 16B alignment)
    *(v8h*)(p)     = __builtin_shufflevector(v, v, 0,1,2,3,4,5,6,7);
    *(v8h*)(p + 8) = __builtin_shufflevector(v, v, 8,9,10,11,12,13,14,15);
}

__device__ __forceinline__ v16h ntx_join(v8h lo, v8h hi) {
    return __builtin_shufflevector(lo, hi, 0,1,2,3,4,5,6,7,8,9,10,11,12,13,14,15);
}

// one K=32 stage: 2 A-frags x 4 B-frags -> 8 WMMAs
__device__ __forceinline__ void ntx_stage(
        const _Float16* __restrict__ Ab, const _Float16* __restrict__ Bb,
        int rg, int cg, int m, int hi, v8f (&acc)[2][4]) {
    #pragma unroll
    for (int rr = 0; rr < 2; ++rr) {
        const _Float16* pa = Ab + (rg * 32 + rr * 16 + m) * LROW + hi * 8;
        const v16h a = ntx_join(*(const v8h*)(pa), *(const v8h*)(pa + 16));
        #pragma unroll
        for (int j = 0; j < 4; ++j) {
            const _Float16* pb = Bb + (cg * 64 + j * 16 + m) * LROW + hi * 8;
            const v16h b = ntx_join(*(const v8h*)(pb), *(const v8h*)(pb + 16));
            acc[rr][j] = __builtin_amdgcn_wmma_f32_16x16x32_f16(
                    false, a, false, b, (short)0, acc[rr][j], false, false);
        }
    }
}

// ---- kernel 3: symmetric WMMA Gram, LDS double-buffered K pipeline ---------
// Grid 64x64, upper triangle only (by >= bx).  Block tile 128x128.
// 8 waves: rg in 0..3 (32-row groups), cg in 0..1 (64-col groups).
__global__ __launch_bounds__(256) void ntx_gram_kernel(
        const _Float16* __restrict__ z, float* __restrict__ denom) {
    const int bx = blockIdx.x, by = blockIdx.y;
    if (by < bx) return;                              // block-uniform exit

    __shared__ _Float16 lds[4 * CH];                  // [buf][A|B][row][k]

    const int tid  = threadIdx.x;
    const int lane = tid & 31;
    const int wave = tid >> 5;
    const int rg = wave & 3, cg = wave >> 2;
    const int m = lane & 15, hi = lane >> 4;

    // cooperative copy: thread t -> row t>>1, 16-half slice (t&1) of the chunk
    const int crow = tid >> 1;
    const int coff = (tid & 1) * 16;
    const _Float16* gA = z + (size_t)(bx * 128 + crow) * DIM + coff;
    const _Float16* gB = z + (size_t)(by * 128 + crow) * DIM + coff;
    const int lOff = crow * LROW + coff;

    v8f acc[2][4];
    #pragma unroll
    for (int rr = 0; rr < 2; ++rr)
        #pragma unroll
        for (int j = 0; j < 4; ++j) acc[rr][j] = (v8f){};

    // prologue: chunk 0 -> buffer 0
    {
        const v16h ra = *(const v16h*)(gA);
        const v16h rb = *(const v16h*)(gB);
        ntx_store16(lds + lOff, ra);
        ntx_store16(lds + CH + lOff, rb);
    }
    __syncthreads();

    for (int s = 0; s < 15; ++s) {
        const int k1 = (s + 1) * KC;
        const v16h ra = *(const v16h*)(gA + k1);      // in flight under WMMAs
        const v16h rb = *(const v16h*)(gB + k1);
        const _Float16* Ab = lds + (s & 1) * 2 * CH;
        ntx_stage(Ab, Ab + CH, rg, cg, m, hi, acc);
        _Float16* An = lds + ((s + 1) & 1) * 2 * CH;
        ntx_store16(An + lOff, ra);
        ntx_store16(An + CH + lOff, rb);
        __syncthreads();
    }
    {   // last stage (chunk 15 sits in buffer 1)
        const _Float16* Ab = lds + 2 * CH;
        ntx_stage(Ab, Ab + CH, rg, cg, m, hi, acc);
    }

    // ---------------- epilogue: exp, diag mask, row + column sums ----------
    // acc[rr][j][v]: row = bx*128 + rg*32 + rr*16 + hi*8 + v
    //               col = by*128 + cg*64 + j*16  + m
    const bool diag = (bx == by);
    #pragma unroll
    for (int rr = 0; rr < 2; ++rr)
        #pragma unroll
        for (int j = 0; j < 4; ++j)
            #pragma unroll
            for (int v = 0; v < 8; ++v) {
                float e = __expf(2.0f * acc[rr][j][v]);
                if (diag && (rg * 32 + rr * 16 + hi * 8 + v ==
                             cg * 64 + j * 16 + m))
                    e = 0.0f;                          // exact diagonal mask
                acc[rr][j][v] = e;
            }

    // row sums over this wave's 64 columns
    #pragma unroll
    for (int rr = 0; rr < 2; ++rr)
        #pragma unroll
        for (int v = 0; v < 8; ++v) {
            float s = (acc[rr][0][v] + acc[rr][1][v]) +
                      (acc[rr][2][v] + acc[rr][3][v]);
            #pragma unroll
            for (int o = 1; o < 16; o <<= 1) s += __shfl_xor(s, o, 32);
            if (m == 0)
                atomicAdd(&denom[bx * 128 + rg * 32 + rr * 16 + hi * 8 + v], s);
        }

    // column sums (transposed contributions) for strict upper-triangle blocks
    if (by > bx) {
        #pragma unroll
        for (int j = 0; j < 4; ++j) {
            float t = 0.0f;
            #pragma unroll
            for (int rr = 0; rr < 2; ++rr)
                #pragma unroll
                for (int v = 0; v < 8; ++v) t += acc[rr][j][v];
            t += __shfl_xor(t, 16, 32);               // combine hi half-groups
            if (hi == 0)
                atomicAdd(&denom[by * 128 + cg * 64 + j * 16 + m], t);
        }
    }
}

// ---------------------- kernel 4: final scalar reduction --------------------
__global__ __launch_bounds__(256) void ntx_finalize_kernel(
        const float* __restrict__ denom, const float* __restrict__ spos,
        float* __restrict__ out) {
    __shared__ float smem[8];
    float s = 0.0f;
    for (int i = threadIdx.x; i < NROWS; i += 256) s += __logf(denom[i]);
    const float slog = block_reduce_sum(s, smem);
    if (threadIdx.x == 0) {
        // loss = (sum log denom - (1/tau)*sum_n pos_n)/(2B); 1/tau = 2,
        // sum_n pos_n = 2*S_pos  ->  subtract 4*S_pos.
        out[0] = (slog - 4.0f * spos[0]) * (1.0f / (float)NROWS);
    }
}

// ---------------------------------------------------------------------------
extern "C" void kernel_launch(void* const* d_in, const int* in_sizes, int n_in,
                              void* d_out, int out_size, void* d_ws, size_t ws_size,
                              hipStream_t stream) {
    const float* ei = (const float*)d_in[0];
    const float* ej = (const float*)d_in[1];
    float* out = (float*)d_out;

    // workspace layout: denom[8192] | S_pos | pad to 8320 floats | z_f16[8192*512]
    float*    denom = (float*)d_ws;
    float*    spos  = denom + NROWS;
    _Float16* z     = (_Float16*)(denom + 8320);   // 33280-byte offset, 64B aligned

    ntx_init_kernel<<<(NROWS + 1 + 255) / 256, 256, 0, stream>>>(denom);
    ntx_normalize_kernel<<<NROWS, 256, 0, stream>>>(ei, ej, z);
    ntx_pos_kernel<<<BROWS, 256, 0, stream>>>(ei, ej, spos);
    dim3 grid(64, 64);
    ntx_gram_kernel<<<grid, 256, 0, stream>>>(z, denom);
    ntx_finalize_kernel<<<1, 256, 0, stream>>>(denom, spos, out);
}
